// MHA_83296595738771
// MI455X (gfx1250) — matrix operand order repaired and checked by reading, not compiled, not run on hardware
//
#include <hip/hip_runtime.h>

typedef __attribute__((ext_vector_type(16))) __bf16 v16bf;
typedef __attribute__((ext_vector_type(8)))  __bf16 v8bf;
typedef __attribute__((ext_vector_type(8)))  float  v8f;

// ---------------------------------------------------------------------------
// WMMA helpers (CDNA5 wave32, v_wmma_f32_16x16x32_bf16)
// ---------------------------------------------------------------------------
static __device__ inline v8f wmma_bf16(v16bf a, v16bf b, v8f c) {
    return __builtin_amdgcn_wmma_f32_16x16x32_bf16(
        /*neg_a=*/false, a, /*neg_b=*/false, b,
        /*c_mod=*/(short)0, c, /*reuse_a=*/false, /*reuse_b=*/false);
}

// A-matrix fragment (16x32 bf16, M x K), source row-major with leading dim ld.
// Lane L: row = row0 + (L&15); lanes 0-15 hold K {0..7,16..23}, lanes 16-31 {8..15,24..31}.
static __device__ inline v16bf load_frag_a(const __bf16* base, int row0, int ld, int k0) {
    int lane  = threadIdx.x & 31;
    int m     = row0 + (lane & 15);
    int khalf = (lane >> 4) * 8;
    const __bf16* p = base + (size_t)m * ld + k0 + khalf;
    v8bf lo = *(const v8bf*)(p);
    v8bf hi = *(const v8bf*)(p + 16);
    return __builtin_shufflevector(lo, hi, 0,1,2,3,4,5,6,7,8,9,10,11,12,13,14,15);
}

// B-matrix fragment (32x16 bf16, K x N) from B^T stored row-major [N][Kld].
// Lane L: col n = n0 + (L&15); lanes 0-15 hold K {0..7,16..23}, lanes 16-31 {8..15,24..31}.
static __device__ inline v16bf load_frag_b(const __bf16* bt, int n0, int ld, int k0) {
    int lane = threadIdx.x & 31;
    int n  = n0 + (lane & 15);
    int kk = k0 + (lane >> 4) * 8;
    const __bf16* p = bt + (size_t)n * ld + kk;
    v8bf lo = *(const v8bf*)(p);
    v8bf hi = *(const v8bf*)(p + 16);
    return __builtin_shufflevector(lo, hi, 0,1,2,3,4,5,6,7,8,9,10,11,12,13,14,15);
}

// ---------------------------------------------------------------------------
// Elementwise conversion / transpose kernels
// ---------------------------------------------------------------------------
__global__ void f32_to_bf16_k(const float* __restrict__ in, __bf16* __restrict__ out, int n) {
    int i = blockIdx.x * blockDim.x + threadIdx.x;
    if (i < n) out[i] = (__bf16)in[i];
}

// in: f32 [K][N] row-major -> out: bf16 [N][K]
__global__ void transpose_bf16_k(const float* __restrict__ in, __bf16* __restrict__ out,
                                 int K, int N) {
    int i = blockIdx.x * blockDim.x + threadIdx.x;
    if (i < K * N) {
        int k = i / N, n = i % N;
        out[(size_t)n * K + k] = (__bf16)in[i];
    }
}

__global__ void init_k(__bf16* Rpad, unsigned int* gmax) {
    int t = threadIdx.x;
    if (t < 64) Rpad[t] = (__bf16)0.0f;   // zero row for rel-shift pad
    if (t == 64) *gmax = 0u;
}

// ---------------------------------------------------------------------------
// Positional basis (Enformer): exp / central-mask / gamma, FS=32 per family
// ---------------------------------------------------------------------------
__global__ void basis_pass1(float* __restrict__ basis, unsigned int* __restrict__ gmax) {
    int i = blockIdx.x;          // 0..2046
    int j = threadIdx.x;         // 0..31
    float pos = (float)(i - 1023);
    float ap  = fabsf(pos);
    float sgn = (pos > 0.0f) ? 1.0f : ((pos < 0.0f) ? -1.0f : 0.0f);
    float* row = basis + (size_t)i * 192;
    // exponential decay: half_life = 2^(3 + 7*j/31)  (log2(1024)=10)
    float hl = exp2f(3.0f + 7.0f * (float)j * (1.0f / 31.0f));
    float fe = __expf(-0.6931471805599453f / hl * ap);
    row[j]      = fe;
    row[32 + j] = sgn * fe;
    // central mask: cw = 2^(j+1)-1
    float cw = exp2f((float)(j + 1)) - 1.0f;
    float fc = (cw > ap) ? 1.0f : 0.0f;
    row[64 + j] = fc;
    row[96 + j] = sgn * fc;
    // gamma pdf: stddev=16, mean linspace(32,1024,32)
    float mean = 32.0f + (1024.0f - 32.0f) * (float)j * (1.0f / 31.0f);
    float conc = (mean / 16.0f) * (mean / 16.0f);
    float rate = mean * (1.0f / 256.0f);
    float xl   = (ap > 0.0f) ? (conc - 1.0f) * __logf(ap) : -3.0e38f;
    float lp   = xl - rate * ap - lgammaf(conc) + conc * __logf(rate);
    float pdf  = __expf(lp) + 1e-8f;
    row[128 + j] = pdf;
    row[160 + j] = sgn * pdf;
    atomicMax(gmax, __float_as_uint(pdf));   // pdf > 0, bit compare == float compare
}

// Normalize gamma block, project with WR -> Rpad (bf16, row i+1), keep R[0] fp32.
__global__ void basis_pass2(const float* __restrict__ basis, const unsigned int* __restrict__ gmax,
                            const float* __restrict__ WR, __bf16* __restrict__ Rpad,
                            float* __restrict__ R0f) {
    int i = blockIdx.x;          // 0..2046
    int d = threadIdx.x;         // 0..63
    float gminv = 1.0f / __uint_as_float(*gmax);
    const float* row = basis + (size_t)i * 192;
    float sum = 0.0f;
    for (int f = 0; f < 192; ++f) {
        float v = row[f];
        if (f >= 128) v *= gminv;
        sum += v * WR[f * 64 + d];
    }
    Rpad[(size_t)(i + 1) * 64 + d] = (__bf16)sum;
    if (i == 0) R0f[d] = sum;
}

// T3[b,h,k] = u[h] . K[b,h,k,:]
__global__ void t3_k(const float* __restrict__ u, const __bf16* __restrict__ Kh,
                     float* __restrict__ T3) {
    int i = blockIdx.x * blockDim.x + threadIdx.x;   // B*H*L = 65536
    if (i >= 65536) return;
    int h = (i >> 10) & 7;
    const __bf16* kp = Kh + (size_t)i * 64;
    float s = 0.0f;
    for (int d = 0; d < 64; ++d) s += u[h * 64 + d] * (float)kp[d];
    T3[i] = s;
}

// c4[h] = v[h] . R[0]
__global__ void c4_k(const float* __restrict__ v, const float* __restrict__ R0f,
                     float* __restrict__ c4) {
    int h = threadIdx.x;
    if (h >= 8) return;
    float s = 0.0f;
    for (int d = 0; d < 64; ++d) s += v[h * 64 + d] * R0f[d];
    c4[h] = s;
}

// ---------------------------------------------------------------------------
// Generic bf16 WMMA GEMM: out = A[M,K] @ B[K,N] + bias, B given transposed [N][K].
// Block = 128M x 128N, 8 waves in a 4x2 grid, each wave = 32x64 (2x4 WMMA tiles).
// mode 0: bf16 row-major [M,N]    mode 1: f32 row-major + bias (final output)
// mode 2: Q/K remap -> [b,h,s,64] mode 3: V remap  -> [b,h,192,s] (transposed)
// ---------------------------------------------------------------------------
__global__ __launch_bounds__(256) void gemm_bf16(
    const __bf16* __restrict__ A, const __bf16* __restrict__ BT,
    const float* __restrict__ bias, __bf16* __restrict__ outb,
    float* __restrict__ outf, int M, int N, int K, int mode) {
    const int w    = threadIdx.x >> 5;
    const int lane = threadIdx.x & 31;
    const int m0   = blockIdx.y * 128 + (w >> 1) * 32;
    const int n0   = blockIdx.x * 128 + (w & 1) * 64;
    const v8f z = {0,0,0,0,0,0,0,0};
    v8f acc[2][4] = {{z, z, z, z}, {z, z, z, z}};
    for (int k0 = 0; k0 < K; k0 += 32) {
        if (k0 + 256 < K) {     // prefetch 8 K-steps ahead (global_prefetch_b8)
            int lm = m0 + (lane & 15), kh = k0 + 256 + ((lane >> 4) << 3);
            __builtin_prefetch(A + (size_t)lm * K + kh, 0, 1);
            int ln = n0 + (lane & 15), kb = k0 + 256 + ((lane >> 4) << 4);
            __builtin_prefetch(BT + (size_t)ln * K + kb, 0, 1);
            __builtin_prefetch(BT + (size_t)(ln + 16) * K + kb, 0, 1);
        }
        v16bf a0 = load_frag_a(A, m0,      K, k0);
        v16bf a1 = load_frag_a(A, m0 + 16, K, k0);
        v16bf b0 = load_frag_b(BT, n0,      K, k0);
        v16bf b1 = load_frag_b(BT, n0 + 16, K, k0);
        v16bf b2 = load_frag_b(BT, n0 + 32, K, k0);
        v16bf b3 = load_frag_b(BT, n0 + 48, K, k0);
        acc[0][0] = wmma_bf16(a0, b0, acc[0][0]);
        acc[0][1] = wmma_bf16(a0, b1, acc[0][1]);
        acc[0][2] = wmma_bf16(a0, b2, acc[0][2]);
        acc[0][3] = wmma_bf16(a0, b3, acc[0][3]);
        acc[1][0] = wmma_bf16(a1, b0, acc[1][0]);
        acc[1][1] = wmma_bf16(a1, b1, acc[1][1]);
        acc[1][2] = wmma_bf16(a1, b2, acc[1][2]);
        acc[1][3] = wmma_bf16(a1, b3, acc[1][3]);
    }
    for (int i = 0; i < 2; ++i)
    for (int j = 0; j < 4; ++j) {
        int rbase = m0 + i * 16 + ((lane >> 4) << 3);
        int col   = n0 + j * 16 + (lane & 15);
        float bv  = bias ? bias[col] : 0.0f;
        for (int r = 0; r < 8; ++r) {
            int row = rbase + r;
            float val = acc[i][j][r] + bv;
            if (mode == 0) {
                outb[(size_t)row * N + col] = (__bf16)val;
            } else if (mode == 1) {
                outf[(size_t)row * N + col] = val;
            } else if (mode == 2) {               // [b,s]x[h,d] -> [b,h,s,d]
                int b = row >> 10, s = row & 1023, h = col >> 6, d = col & 63;
                outb[(((size_t)(b * 8 + h) * 1024 + s) << 6) + d] = (__bf16)val;
            } else {                              // [b,s]x[h,vd] -> [b,h,vd,s]
                int b = row >> 10, s = row & 1023, h = col / 192, vd = col % 192;
                outb[((size_t)(b * 8 + h) * 192 + vd) * 1024 + s] = (__bf16)val;
            }
        }
    }
}

// ---------------------------------------------------------------------------
// Flash-style relative attention. Block = (b,h, 64-query tile), 256 threads.
// ---------------------------------------------------------------------------
__global__ __launch_bounds__(256) void attn_k(
    const __bf16* __restrict__ Qh, const __bf16* __restrict__ Kh,
    const __bf16* __restrict__ Vt, const __bf16* __restrict__ Rpad,
    const float* __restrict__ T3, const float* __restrict__ c4,
    __bf16* __restrict__ Oh) {
    const int bh = blockIdx.y;
    const int b  = bh >> 3, h = bh & 7;
    const int q0 = blockIdx.x * 64;
    const int tid = threadIdx.x, w = tid >> 5, lane = tid & 31;
    const __bf16* Qb = Qh + (size_t)bh * 1024 * 64;
    const __bf16* Kb = Kh + (size_t)bh * 1024 * 64;
    const __bf16* Vb = Vt + (size_t)bh * 192 * 1024;
    const float* T3b = T3 + (size_t)bh * 1024;
    const float  c4h = c4[h];

    __shared__ float S1[64][65];
    __shared__ float SE[32][65];
    __shared__ float SO[32][65];
    __shared__ alignas(16) __bf16 P[64][64];
    __shared__ float csc[64], linv[64];

    const v8f z = {0,0,0,0,0,0,0,0};
    v8f o[6] = {z, z, z, z, z, z};
    float mrow = -3.0e38f, lrow = 0.0f;

    const int qsub = w >> 1;     // output row tile owned by this wave
    const int vh   = w & 1;      // output col half (96 cols)

    for (int k0 = 0; k0 < 1024; k0 += 64) {
        __syncthreads();
        // ---- phase 1: scores by WMMA into LDS -----------------------------
        {   // S1 = Q . K^T : 16 tiles, 2 per wave
            int qt = w >> 1;
            for (int jj = 0; jj < 2; ++jj) {
                int kt = (w & 1) * 2 + jj;
                v8f acc = z;
                for (int kk = 0; kk < 64; kk += 32) {
                    v16bf a  = load_frag_a(Qb, q0 + qt * 16, 64, kk);
                    v16bf bb = load_frag_b(Kb, k0 + kt * 16, 64, kk);
                    acc = wmma_bf16(a, bb, acc);
                }
                int r0 = qt * 16 + ((lane >> 4) << 3), cc = kt * 16 + (lane & 15);
                for (int r = 0; r < 8; ++r) S1[r0 + r][cc] = acc[r];
            }
            int qt2 = w >> 2, kt = w & 3;
            {   // QRE: rows q0/2.., window Rpad[1024+k0 ..]   (even query rows)
                v8f acc = z;
                for (int kk = 0; kk < 64; kk += 32) {
                    v16bf a  = load_frag_a(Qb, q0 / 2 + qt2 * 16, 64, kk);
                    v16bf bb = load_frag_b(Rpad, 1024 + k0 + kt * 16, 64, kk);
                    acc = wmma_bf16(a, bb, acc);
                }
                int r0 = qt2 * 16 + ((lane >> 4) << 3), cc = kt * 16 + (lane & 15);
                for (int r = 0; r < 8; ++r) SE[r0 + r][cc] = acc[r];
            }
            {   // QRO: rows q0/2+1.., window Rpad[k0 ..]      (odd query rows)
                v8f acc = z;
                for (int kk = 0; kk < 64; kk += 32) {
                    v16bf a  = load_frag_a(Qb, q0 / 2 + 1 + qt2 * 16, 64, kk);
                    v16bf bb = load_frag_b(Rpad, k0 + kt * 16, 64, kk);
                    acc = wmma_bf16(a, bb, acc);
                }
                int r0 = qt2 * 16 + ((lane >> 4) << 3), cc = kt * 16 + (lane & 15);
                for (int r = 0; r < 8; ++r) SO[r0 + r][cc] = acc[r];
            }
        }
        __syncthreads();
        // ---- phase 2: online softmax (one thread per query row) -----------
        if (tid < 64) {
            int t = tid;
            float rmax = -3.0e38f;
            for (int k = 0; k < 64; ++k) {
                float t2 = (t & 1) ? SO[(t - 1) >> 1][k] : SE[t >> 1][k];
                float s  = (S1[t][k] + t2 + T3b[k0 + k] + c4h) * 0.125f;
                rmax = fmaxf(rmax, s);
            }
            float mnew = fmaxf(mrow, rmax);
            float c    = __expf(mrow - mnew);
            float lsum = 0.0f;
            for (int k = 0; k < 64; ++k) {
                float t2 = (t & 1) ? SO[(t - 1) >> 1][k] : SE[t >> 1][k];
                float s  = (S1[t][k] + t2 + T3b[k0 + k] + c4h) * 0.125f;
                float p  = __expf(s - mnew);
                P[t][k]  = (__bf16)p;
                lsum += p;
            }
            lrow = lrow * c + lsum;
            mrow = mnew;
            csc[t] = c;
        }
        __syncthreads();
        // ---- phase 3: rescale accumulators, O += P @ V --------------------
        {
            int r0 = qsub * 16 + ((lane >> 4) << 3);
            for (int j = 0; j < 6; ++j)
                for (int r = 0; r < 8; ++r) o[j][r] *= csc[r0 + r];
            for (int kk = 0; kk < 64; kk += 32) {
                v16bf a = load_frag_a((const __bf16*)&P[0][0], qsub * 16, 64, kk);
                for (int j = 0; j < 6; ++j) {
                    v16bf bb = load_frag_b(Vb, vh * 96 + j * 16, 1024, k0 + kk);
                    o[j] = wmma_bf16(a, bb, o[j]);
                }
            }
        }
    }
    if (tid < 64) linv[tid] = 1.0f / lrow;
    __syncthreads();
    {   // epilogue: normalize and write Oh[b, q, h*192 + col] (bf16 [8192][1536])
        int r0 = qsub * 16 + ((lane >> 4) << 3);
        for (int j = 0; j < 6; ++j) {
            int col = h * 192 + vh * 96 + j * 16 + (lane & 15);
            for (int r = 0; r < 8; ++r) {
                int row = q0 + r0 + r;
                float val = o[j][r] * linv[r0 + r];
                Oh[((size_t)b * 1024 + row) * 1536 + col] = (__bf16)val;
            }
        }
    }
}

// ---------------------------------------------------------------------------
// Launch
// ---------------------------------------------------------------------------
extern "C" void kernel_launch(void* const* d_in, const int* in_sizes, int n_in,
                              void* d_out, int out_size, void* d_ws, size_t ws_size,
                              hipStream_t stream) {
    const float* x  = (const float*)d_in[0];
    const float* Wq = (const float*)d_in[1];
    const float* bq = (const float*)d_in[2];
    const float* Wk = (const float*)d_in[3];
    const float* bk = (const float*)d_in[4];
    const float* Wv = (const float*)d_in[5];
    const float* bv = (const float*)d_in[6];
    const float* Wo = (const float*)d_in[7];
    const float* bo = (const float*)d_in[8];
    const float* WR = (const float*)d_in[9];
    const float* u  = (const float*)d_in[10];
    const float* v  = (const float*)d_in[11];
    float* out = (float*)d_out;

    char* ws = (char*)d_ws;
    size_t off = 0;
    auto alloc = [&](size_t bytes) -> char* {
        char* p = ws + off;
        off = (off + bytes + 255) & ~(size_t)255;
        return p;
    };
    __bf16* xh    = (__bf16*)alloc((size_t)8192 * 1536 * 2);
    __bf16* WqT   = (__bf16*)alloc((size_t)512 * 1536 * 2);
    __bf16* WkT   = (__bf16*)alloc((size_t)512 * 1536 * 2);
    __bf16* WvT   = (__bf16*)alloc((size_t)1536 * 1536 * 2);
    __bf16* WoT   = (__bf16*)alloc((size_t)1536 * 1536 * 2);
    __bf16* Qhb   = (__bf16*)alloc((size_t)64 * 1024 * 64 * 2);
    __bf16* Khb   = (__bf16*)alloc((size_t)64 * 1024 * 64 * 2);
    __bf16* Vtb   = (__bf16*)alloc((size_t)64 * 192 * 1024 * 2);
    __bf16* Rpad  = (__bf16*)alloc((size_t)2048 * 64 * 2);
    float*  basis = (float*) alloc((size_t)2047 * 192 * 4);
    float*  R0f   = (float*) alloc(64 * 4);
    float*  c4    = (float*) alloc(8 * 4);
    unsigned int* gmax = (unsigned int*)alloc(4);
    float*  T3    = (float*) alloc((size_t)65536 * 4);
    __bf16* Oh    = (__bf16*)alloc((size_t)8192 * 1536 * 2);

    // 1) conversions / transposes
    f32_to_bf16_k<<<(8192 * 1536) / 256, 256, 0, stream>>>(x, xh, 8192 * 1536);
    transpose_bf16_k<<<(1536 * 512 + 255) / 256, 256, 0, stream>>>(Wq, WqT, 1536, 512);
    transpose_bf16_k<<<(1536 * 512 + 255) / 256, 256, 0, stream>>>(Wk, WkT, 1536, 512);
    transpose_bf16_k<<<(1536 * 1536 + 255) / 256, 256, 0, stream>>>(Wv, WvT, 1536, 1536);
    transpose_bf16_k<<<(1536 * 1536 + 255) / 256, 256, 0, stream>>>(Wo, WoT, 1536, 1536);

    // 2) positional basis -> Rpad
    init_k<<<1, 128, 0, stream>>>(Rpad, gmax);
    basis_pass1<<<2047, 32, 0, stream>>>(basis, gmax);
    basis_pass2<<<2047, 64, 0, stream>>>(basis, gmax, WR, Rpad, R0f);

    // 3) Q/K/V projections (WMMA GEMMs), block tile 128x128
    gemm_bf16<<<dim3(4, 64), 256, 0, stream>>>(xh, WqT, bq, Qhb, nullptr, 8192, 512, 1536, 2);
    gemm_bf16<<<dim3(4, 64), 256, 0, stream>>>(xh, WkT, bk, Khb, nullptr, 8192, 512, 1536, 2);
    gemm_bf16<<<dim3(12, 64), 256, 0, stream>>>(xh, WvT, bv, Vtb, nullptr, 8192, 1536, 1536, 3);

    // 4) per-column bias term and per-head scalar
    t3_k<<<65536 / 256, 256, 0, stream>>>(u, Khb, T3);
    c4_k<<<1, 8, 0, stream>>>(v, R0f, c4);

    // 5) relative attention
    attn_k<<<dim3(16, 64), 256, 0, stream>>>(Qhb, Khb, Vtb, Rpad, T3, c4, Oh);

    // 6) output projection -> fp32 d_out
    gemm_bf16<<<dim3(12, 64), 256, 0, stream>>>(Oh, WoT, bo, nullptr, out, 8192, 1536, 1536, 1);
}
